// ResidualMixedMambaBlock_10127532884274
// MI455X (gfx1250) — matrix operand order repaired and checked by reading, not compiled, not run on hardware
//
#include <hip/hip_runtime.h>
#include <hip/hip_bf16.h>

// ---------------------------------------------------------------------------
// ResidualMixedMambaBlock for MI455X (gfx1250, wave32, WMMA + TDM)
//   B=8, EMBED=192, D_INNER=384, D_STATE=4, D_CONV=4, DT_RANK=12, L=4096
// ---------------------------------------------------------------------------

typedef __attribute__((ext_vector_type(16))) _Float16 v16h;
typedef __attribute__((ext_vector_type(8)))  _Float16 v8h;
typedef __attribute__((ext_vector_type(8)))  float    v8f;
typedef __attribute__((ext_vector_type(4)))  unsigned int u32x4;
typedef __attribute__((ext_vector_type(8)))  int      i32x8;
typedef __attribute__((ext_vector_type(4)))  int      i32x4;

#define B_N     8
#define L_N     4096
#define M_N     (B_N * L_N)     // 32768 rows
#define EMBED_N 192
#define DI_N    384             // D_INNER
#define DS_N    4               // D_STATE
#define DTR_N   12              // DT_RANK
#define XDBL_P  32              // padded (DT_RANK + 2*D_STATE)=20 -> 32
#define SEG_N   64              // scan segments
#define SEGLEN  64              // steps per segment (64*64 = 4096)

#if defined(__gfx1250__) && __has_builtin(__builtin_amdgcn_tensor_load_to_lds) && \
    __has_builtin(__builtin_amdgcn_s_wait_tensorcnt)
#define USE_TDM 1
#else
#define USE_TDM 0
#endif

__device__ __forceinline__ float siluf(float v)   { return v / (1.0f + __expf(-v)); }
__device__ __forceinline__ float softpf(float v)  { return v > 20.0f ? v : log1pf(__expf(v)); }

// ---------------------------------------------------------------------------
// Transpose x (B,C,L) f32  ->  xres (B,L,C) f32  and  h16 (B,L,C) f16
// ---------------------------------------------------------------------------
__global__ __launch_bounds__(256) void transpose_in(
    const float* __restrict__ x, float* __restrict__ xres, _Float16* __restrict__ h16) {
  __shared__ float tile[32][33];
  int b  = blockIdx.z;
  int l0 = blockIdx.x * 32;
  int c0 = blockIdx.y * 32;
  int tx = threadIdx.x, ty = threadIdx.y;
  const float* xb = x + (size_t)b * EMBED_N * L_N;
#pragma unroll
  for (int i = 0; i < 4; i++) {
    int c = c0 + ty + i * 8;
    tile[ty + i * 8][tx] = xb[(size_t)c * L_N + l0 + tx];   // coalesced over l
  }
  __syncthreads();
#pragma unroll
  for (int i = 0; i < 4; i++) {
    int l = l0 + ty + i * 8;
    float v = tile[tx][ty + i * 8];
    size_t o = ((size_t)b * L_N + l) * EMBED_N + c0 + tx;   // coalesced over c
    xres[o] = v;
    h16[o]  = (_Float16)v;
  }
}

// ---------------------------------------------------------------------------
// Transpose lnout (B,L,C) f32 -> out (B,C,L) f32
// ---------------------------------------------------------------------------
__global__ __launch_bounds__(256) void transpose_out(
    const float* __restrict__ ln, float* __restrict__ out) {
  __shared__ float tile[32][33];
  int b  = blockIdx.z;
  int l0 = blockIdx.x * 32;
  int c0 = blockIdx.y * 32;
  int tx = threadIdx.x, ty = threadIdx.y;
#pragma unroll
  for (int i = 0; i < 4; i++) {
    int l = l0 + ty + i * 8;
    tile[ty + i * 8][tx] = ln[((size_t)b * L_N + l) * EMBED_N + c0 + tx];
  }
  __syncthreads();
#pragma unroll
  for (int i = 0; i < 4; i++) {
    int c = c0 + ty + i * 8;
    out[((size_t)b * EMBED_N + c) * L_N + l0 + tx] = tile[tx][ty + i * 8];
  }
}

// ---------------------------------------------------------------------------
// f32 -> f16 convert with zero padding (generic; also plain convert)
// ---------------------------------------------------------------------------
__global__ void cvt_pad_f16(const float* __restrict__ src, _Float16* __restrict__ dst,
                            int srows, int scols, int drows, int dcols) {
  long i = (long)blockIdx.x * 256 + threadIdx.x;
  long n = (long)drows * dcols;
  if (i >= n) return;
  int r = (int)(i / dcols), c = (int)(i % dcols);
  dst[i] = (r < srows && c < scols) ? (_Float16)src[(size_t)r * scols + c] : (_Float16)0.0f;
}

// ---------------------------------------------------------------------------
// WMMA GEMM:  C[m][n] = sum_k A[m][k] * W[n][k]      (A: MxK f16, W: NxK f16)
// Block = 8 waves = 128 x (16*NT) output tile. The (16*NT) x K weight tile is
// staged once per block into LDS via the Tensor Data Mover (TDM descriptor,
// count=1, 2D tile: dim0=K elems of 2B, dim1=16*NT rows), then all 8 waves
// read B fragments from LDS while streaming A from global.
// Fragments (A/B: contiguous k order used here, not the ISA table; C per ISA 7.12.2):
//   A 16x32 f16 : lane = 16*half + m ; elems e    -> K=half*16+e
//   B 32x16 f16 : lane = 16*half + n ; elems e    -> K=half*16+e
//   C 16x16 f32 : lane = 16*half + n ; elem r     -> M=r+half*8
// ---------------------------------------------------------------------------
template <int NT>
__global__ __launch_bounds__(256) void gemm_f16_wmma(
    const _Float16* __restrict__ A, const _Float16* __restrict__ W,
    float* __restrict__ C, int M, int N, int K) {
  __shared__ __align__(128) _Float16 wtile[64 * 384];   // up to 48 KB (<320 KB WGP LDS)

  int tid  = threadIdx.x;
  int lane = tid & 31;
  int nCols  = N / (16 * NT);
  int blockM = blockIdx.x / nCols;
  int nBase  = (blockIdx.x % nCols) * (16 * NT);
  int mTile  = blockM * 8 + (tid >> 5);
  int half   = lane >> 4;
  int sub    = lane & 15;
  int rows   = 16 * NT;

#if USE_TDM
  if (tid < 32) {                      // wave 0 issues one TDM DMA for the block
    unsigned long long ga = (unsigned long long)(const void*)(W + (size_t)nBase * K);
    u32x4 g0;
    g0[0] = 1u;                                        // count=1, user mode
    g0[1] = 0u;                                        // lds_addr: wtile is LDS offset 0
    g0[2] = (unsigned int)ga;                          // global_addr[31:0]
    g0[3] = (unsigned int)((ga >> 32) & 0x1FFFFFFu) | (2u << 30);  // addr[56:32] | type=2
    unsigned int dim0 = (unsigned int)K;               // elements per row
    unsigned int dim1 = (unsigned int)rows;            // rows in tile
    unsigned long long st0 = (unsigned long long)K;    // row stride (elements)
    unsigned long long st1 = (unsigned long long)K;
    i32x8 g1;
    g1[0] = (int)(1u << 16);                           // wg_mask=0, data_size=1 (2 bytes)
    g1[1] = (int)((dim0 & 0xFFFFu) << 16);             // tensor_dim0[15:0]
    g1[2] = (int)(((dim0 >> 16) & 0xFFFFu) | ((dim1 & 0xFFFFu) << 16));
    g1[3] = (int)(((dim1 >> 16) & 0xFFFFu) | ((dim0 & 0xFFFFu) << 16)); // tile_dim0=K
    g1[4] = (int)(dim1 & 0xFFFFu);                     // tile_dim1=rows, tile_dim2=0
    g1[5] = (int)(unsigned int)(st0 & 0xFFFFFFFFu);    // tensor_dim0_stride[31:0]
    g1[6] = (int)(((st0 >> 32) & 0xFFFFu) | ((st1 & 0xFFFFu) << 16));
    g1[7] = (int)(unsigned int)(st1 >> 16);
    i32x4 z4 = {0, 0, 0, 0};
    i32x8 z8 = {0, 0, 0, 0, 0, 0, 0, 0};
    __builtin_amdgcn_tensor_load_to_lds(g0, g1, z4, z4, z8, 0);   // 6-arg (clang-23)
    __builtin_amdgcn_s_wait_tensorcnt(0);
  }
#else
  {                                    // cooperative fallback: 16B per thread per step
    int total = rows * K / 8;          // v8h chunks
    for (int i = tid; i < total; i += 256)
      *(v8h*)(wtile + (size_t)i * 8) = *(const v8h*)(W + (size_t)nBase * K + (size_t)i * 8);
  }
#endif
  __syncthreads();

  const _Float16* Arow = A + (size_t)(mTile * 16 + sub) * K;

  v8f acc[NT];
  const v8f vzero = {0.f, 0.f, 0.f, 0.f, 0.f, 0.f, 0.f, 0.f};
#pragma unroll
  for (int t = 0; t < NT; t++) acc[t] = vzero;

  for (int k0 = 0; k0 < K; k0 += 32) {
    v8h alo = *(const v8h*)(Arow + k0 + half * 16);
    v8h ahi = *(const v8h*)(Arow + k0 + half * 16 + 8);
    v16h a;
#pragma unroll
    for (int i = 0; i < 8; i++) { a[i] = alo[i]; a[i + 8] = ahi[i]; }
#pragma unroll
    for (int t = 0; t < NT; t++) {
      v16h bfr = *(const v16h*)(wtile + (size_t)(t * 16 + sub) * K + k0 + half * 16);
      acc[t] = __builtin_amdgcn_wmma_f32_16x16x32_f16(
          false, a, false, bfr, (short)0, acc[t], false, false);
    }
  }
#pragma unroll
  for (int t = 0; t < NT; t++) {
#pragma unroll
    for (int r = 0; r < 8; r++) {
      int m = mTile * 16 + r + half * 8;
      int n = nBase + t * 16 + sub;
      C[(size_t)m * N + n] = acc[t][r];
    }
  }
}

// ---------------------------------------------------------------------------
// Depthwise causal conv (width 4) + bias + SiLU.
// xin = xz[:, 0:384] (row stride 768). Writes xa f32 and xa16 f16.
// ---------------------------------------------------------------------------
__global__ __launch_bounds__(128) void conv_silu(
    const float* __restrict__ xz, const float* __restrict__ cw,
    const float* __restrict__ cb, float* __restrict__ xa, _Float16* __restrict__ xa16) {
  int d = blockIdx.x * 128 + threadIdx.x;   // 0..383
  int l = blockIdx.y;
  int b = blockIdx.z;
  float acc = cb[d];
#pragma unroll
  for (int j = 0; j < 4; j++) {
    int ls = l - 3 + j;
    if (ls >= 0)
      acc += xz[((size_t)b * L_N + ls) * (2 * DI_N) + d] * cw[d * 4 + j];
  }
  float s = siluf(acc);
  size_t row = (size_t)b * L_N + l;
  xa[row * DI_N + d]   = s;
  xa16[row * DI_N + d] = (_Float16)s;
}

// ---------------------------------------------------------------------------
// Segmented selective scan (h = dA*h + dt*B*xa is associative in (a,b) form)
// Pass 1: per (b,g,d,s) compute segment summary (prod dA, local h from 0)
// ---------------------------------------------------------------------------
__global__ __launch_bounds__(256) void scan_pass1(
    const float* __restrict__ dtraw, const float* __restrict__ xa,
    const float* __restrict__ xdbl, const float* __restrict__ dtb,
    const float* __restrict__ Alog, float* __restrict__ Aseg, float* __restrict__ Bseg) {
  long idx = (long)blockIdx.x * 256 + threadIdx.x;   // (b*64+g)*384*4 + d*4 + s
  int s = (int)(idx & 3);
  long t = idx >> 2;
  int d = (int)(t % DI_N); t /= DI_N;
  int g = (int)(t % SEG_N);
  int b = (int)(t / SEG_N);
  float As   = -__expf(Alog[d * DS_N + s]);
  float bias = dtb[d];
  float aprod = 1.0f, h = 0.0f;
  long lbase = (long)b * L_N + g * SEGLEN;
  for (int k = 0; k < SEGLEN; k++) {
    long row = lbase + k;
    float dtv = softpf(dtraw[row * DI_N + d] + bias);
    float xav = xa[row * DI_N + d];
    float Bv  = xdbl[row * XDBL_P + DTR_N + s];
    float dA  = __expf(dtv * As);
    h = dA * h + dtv * Bv * xav;
    aprod *= dA;
  }
  Aseg[idx] = aprod;
  Bseg[idx] = h;
}

// Pass 2: chain 64 segment summaries per (b,d,s); store each segment's h_in.
__global__ __launch_bounds__(256) void scan_pass2(
    const float* __restrict__ Aseg, const float* __restrict__ Bseg,
    float* __restrict__ Hstart) {
  long idx = (long)blockIdx.x * 256 + threadIdx.x;   // 8*384*4 = 12288
  int s = (int)(idx & 3);
  int d = (int)((idx >> 2) % DI_N);
  int b = (int)(idx / (DI_N * DS_N));
  float h = 0.0f;
  for (int g = 0; g < SEG_N; g++) {
    long o = ((((long)b * SEG_N + g) * DI_N) + d) * DS_N + s;
    Hstart[o] = h;
    h = Aseg[o] * h + Bseg[o];
  }
}

// Pass 3: replay each segment from Hstart, emit gated output y16 (f16).
__global__ __launch_bounds__(256) void scan_pass3(
    const float* __restrict__ dtraw, const float* __restrict__ xa,
    const float* __restrict__ xdbl, const float* __restrict__ xz,
    const float* __restrict__ dtb, const float* __restrict__ Alog,
    const float* __restrict__ Dw, const float* __restrict__ Hstart,
    _Float16* __restrict__ y16) {
  long idx = (long)blockIdx.x * 256 + threadIdx.x;   // (b*64+g)*384 + d
  int d = (int)(idx % DI_N);
  int g = (int)((idx / DI_N) % SEG_N);
  int b = (int)(idx / (DI_N * SEG_N));
  float As[DS_N], h[DS_N];
  long segoff = ((((long)b * SEG_N + g) * DI_N) + d) * DS_N;
#pragma unroll
  for (int s = 0; s < DS_N; s++) {
    As[s] = -__expf(Alog[d * DS_N + s]);
    h[s]  = Hstart[segoff + s];
  }
  float bias = dtb[d], Dv = Dw[d];
  long lbase = (long)b * L_N + g * SEGLEN;
  for (int k = 0; k < SEGLEN; k++) {
    long row = lbase + k;
    if (k + 8 < SEGLEN)  // gfx1250 global_prefetch_b8
      __builtin_prefetch(dtraw + (row + 8) * DI_N + d, 0, 1);
    float dtv = softpf(dtraw[row * DI_N + d] + bias);
    float xav = xa[row * DI_N + d];
    float y = 0.0f;
#pragma unroll
    for (int s = 0; s < DS_N; s++) {
      float Bv = xdbl[row * XDBL_P + DTR_N + s];
      float Cv = xdbl[row * XDBL_P + DTR_N + DS_N + s];
      float dA = __expf(dtv * As[s]);
      h[s] = dA * h[s] + dtv * Bv * xav;
      y += h[s] * Cv;
    }
    y += xav * Dv;
    float zv = xz[row * (2 * DI_N) + DI_N + d];
    y *= siluf(zv);
    y16[row * DI_N + d] = (_Float16)y;
  }
}

// ---------------------------------------------------------------------------
// Fused rank-4 projection + residual + LayerNorm over C=192.
// One wave per (b,l) row; 6 channels per lane; wave32 shuffle reductions.
// ---------------------------------------------------------------------------
__global__ __launch_bounds__(256) void lowrank_ln(
    const float* __restrict__ h, const float* __restrict__ xres,
    const float* __restrict__ Uw, const float* __restrict__ Vw,
    const float* __restrict__ lng, const float* __restrict__ lnb,
    float* __restrict__ out) {
  int row  = blockIdx.x * 8 + (threadIdx.x >> 5);
  int lane = threadIdx.x & 31;
  const float* hr = h + (size_t)row * EMBED_N;
  float hv[6];
#pragma unroll
  for (int j = 0; j < 6; j++) hv[j] = hr[lane * 6 + j];
  float tr[4];
#pragma unroll
  for (int r = 0; r < 4; r++) {
    float p = 0.0f;
#pragma unroll
    for (int j = 0; j < 6; j++) p += hv[j] * Uw[r * EMBED_N + lane * 6 + j];
#pragma unroll
    for (int m = 16; m >= 1; m >>= 1) p += __shfl_xor(p, m, 32);
    tr[r] = p;
  }
  const float* xr = xres + (size_t)row * EMBED_N;
  float val[6], lsum = 0.0f, lsq = 0.0f;
#pragma unroll
  for (int j = 0; j < 6; j++) {
    int c = lane * 6 + j;
    float v = tr[0] * Vw[c * 4 + 0] + tr[1] * Vw[c * 4 + 1] +
              tr[2] * Vw[c * 4 + 2] + tr[3] * Vw[c * 4 + 3] + xr[c];
    val[j] = v; lsum += v; lsq += v * v;
  }
#pragma unroll
  for (int m = 16; m >= 1; m >>= 1) {
    lsum += __shfl_xor(lsum, m, 32);
    lsq  += __shfl_xor(lsq,  m, 32);
  }
  float mu  = lsum * (1.0f / EMBED_N);
  float var = lsq * (1.0f / EMBED_N) - mu * mu;
  float rs  = rsqrtf(var + 1e-5f);
#pragma unroll
  for (int j = 0; j < 6; j++) {
    int c = lane * 6 + j;
    out[(size_t)row * EMBED_N + c] = (val[j] - mu) * rs * lng[c] + lnb[c];
  }
}

// ---------------------------------------------------------------------------
// Host orchestration
// ---------------------------------------------------------------------------
static inline size_t alignup(size_t v) { return (v + 255) & ~(size_t)255; }

extern "C" void kernel_launch(void* const* d_in, const int* in_sizes, int n_in,
                              void* d_out, int out_size, void* d_ws, size_t ws_size,
                              hipStream_t stream) {
  const float* x        = (const float*)d_in[0];
  const float* in_projw = (const float*)d_in[1];
  const float* conv_w   = (const float*)d_in[2];
  const float* conv_b   = (const float*)d_in[3];
  const float* x_projw  = (const float*)d_in[4];
  const float* dt_projw = (const float*)d_in[5];
  const float* dt_projb = (const float*)d_in[6];
  const float* A_log    = (const float*)d_in[7];
  const float* Dw       = (const float*)d_in[8];
  const float* out_projw= (const float*)d_in[9];
  const float* U_w      = (const float*)d_in[10];
  const float* V_w      = (const float*)d_in[11];
  const float* ln_g     = (const float*)d_in[12];
  const float* ln_b     = (const float*)d_in[13];
  float* outp = (float*)d_out;

  char* ws = (char*)d_ws;
  size_t off = 0;
  auto take = [&](size_t bytes) { char* p = ws + off; off += alignup(bytes); return p; };

  _Float16* h16     = (_Float16*)take((size_t)M_N * EMBED_N * 2);
  float*    xres    = (float*)   take((size_t)M_N * EMBED_N * 4);
  float*    xz      = (float*)   take((size_t)M_N * 2 * DI_N * 4);
  float*    xa      = (float*)   take((size_t)M_N * DI_N * 4);
  _Float16* xa16    = (_Float16*)take((size_t)M_N * DI_N * 2);
  float*    xdbl    = (float*)   take((size_t)M_N * XDBL_P * 4);
  _Float16* xdbl16  = (_Float16*)take((size_t)M_N * XDBL_P * 2);
  float*    dtraw   = (float*)   take((size_t)M_N * DI_N * 4);
  _Float16* y16     = (_Float16*)take((size_t)M_N * DI_N * 2);
  float*    hbuf    = (float*)   take((size_t)M_N * EMBED_N * 4);
  float*    Aseg    = (float*)   take((size_t)B_N * SEG_N * DI_N * DS_N * 4);
  float*    Bseg    = (float*)   take((size_t)B_N * SEG_N * DI_N * DS_N * 4);
  float*    Hstart  = (float*)   take((size_t)B_N * SEG_N * DI_N * DS_N * 4);
  _Float16* w_in16  = (_Float16*)take((size_t)2 * DI_N * EMBED_N * 2);
  _Float16* w_xp16  = (_Float16*)take((size_t)XDBL_P * DI_N * 2);
  _Float16* w_dt16  = (_Float16*)take((size_t)DI_N * XDBL_P * 2);
  _Float16* w_out16 = (_Float16*)take((size_t)EMBED_N * DI_N * 2);

  auto cvt = [&](const float* s, _Float16* d, int sr, int sc, int dr, int dc) {
    long n = (long)dr * dc;
    cvt_pad_f16<<<(unsigned)((n + 255) / 256), 256, 0, stream>>>(s, d, sr, sc, dr, dc);
  };

  // x (B,C,L) -> xres (B,L,C) f32 + h16 f16
  transpose_in<<<dim3(L_N / 32, EMBED_N / 32, B_N), dim3(32, 8), 0, stream>>>(x, xres, h16);

  for (int mix = 0; mix < 2; mix++) {
    const float* inw = in_projw  + (size_t)mix * 2 * DI_N * EMBED_N;
    const float* cw  = conv_w    + (size_t)mix * DI_N * 4;
    const float* cb  = conv_b    + (size_t)mix * DI_N;
    const float* xpw = x_projw   + (size_t)mix * 20 * DI_N;
    const float* dtw = dt_projw  + (size_t)mix * DI_N * DTR_N;
    const float* dtb = dt_projb  + (size_t)mix * DI_N;
    const float* Al  = A_log     + (size_t)mix * DI_N * DS_N;
    const float* Dv  = Dw        + (size_t)mix * DI_N;
    const float* opw = out_projw + (size_t)mix * EMBED_N * DI_N;

    if (mix > 0) cvt(hbuf, h16, M_N, EMBED_N, M_N, EMBED_N);
    cvt(inw, w_in16, 2 * DI_N, EMBED_N, 2 * DI_N, EMBED_N);      // 768 x 192
    cvt(xpw, w_xp16, 20, DI_N, XDBL_P, DI_N);                    // pad N 20 -> 32
    cvt(dtw, w_dt16, DI_N, DTR_N, DI_N, XDBL_P);                 // pad K 12 -> 32
    cvt(opw, w_out16, EMBED_N, DI_N, EMBED_N, DI_N);             // 192 x 384

    // in_proj: xz (M,768) = h16 (M,192) x w_in16^T   (block = 128x64 tile)
    gemm_f16_wmma<4><<<(M_N / 128) * (2 * DI_N / 64), 256, 0, stream>>>(
        h16, w_in16, xz, M_N, 2 * DI_N, EMBED_N);

    // depthwise conv + SiLU -> xa / xa16
    conv_silu<<<dim3(DI_N / 128, L_N, B_N), 128, 0, stream>>>(xz, cw, cb, xa, xa16);

    // x_proj: xdbl (M,32) = xa16 (M,384) x w_xp16^T   (rows 20..31 of W are 0)
    gemm_f16_wmma<2><<<(M_N / 128) * (XDBL_P / 32), 256, 0, stream>>>(
        xa16, w_xp16, xdbl, M_N, XDBL_P, DI_N);
    cvt(xdbl, xdbl16, M_N, XDBL_P, M_N, XDBL_P);

    // dt_proj: dtraw (M,384) = xdbl16 (M,32) x w_dt16^T (cols 12..31 of W are 0)
    gemm_f16_wmma<4><<<(M_N / 128) * (DI_N / 64), 256, 0, stream>>>(
        xdbl16, w_dt16, dtraw, M_N, DI_N, XDBL_P);

    // segmented selective scan (bias+softplus fused; gate with silu(z))
    scan_pass1<<<(B_N * SEG_N * DI_N * DS_N) / 256, 256, 0, stream>>>(
        dtraw, xa, xdbl, dtb, Al, Aseg, Bseg);
    scan_pass2<<<(B_N * DI_N * DS_N) / 256, 256, 0, stream>>>(Aseg, Bseg, Hstart);
    scan_pass3<<<(B_N * SEG_N * DI_N) / 256, 256, 0, stream>>>(
        dtraw, xa, xdbl, xz, dtb, Al, Dv, Hstart, y16);

    // out_proj: hbuf (M,192) = y16 (M,384) x w_out16^T
    gemm_f16_wmma<4><<<(M_N / 128) * (EMBED_N / 64), 256, 0, stream>>>(
        y16, w_out16, hbuf, M_N, EMBED_N, DI_N);
  }

  // rank-4 U/V + residual + LayerNorm -> (B,L,C) in xa buffer (free by now)
  float* lnout = xa;
  lowrank_ln<<<M_N / 8, 256, 0, stream>>>(hbuf, xres, U_w, V_w, ln_g, ln_b, lnout);

  // (B,L,C) -> (B,C,H,W)
  transpose_out<<<dim3(L_N / 32, EMBED_N / 32, B_N), dim3(32, 8), 0, stream>>>(lnout, outp);
}
